// LSTM2Layer_62354335203952
// MI455X (gfx1250) — compile-verified, matrix operand order checked by reading, not run on hardware
//
#include <hip/hip_runtime.h>

namespace {

constexpr int kH    = 32;    // LSTM hidden
constexpr int kT    = 1024;  // sequence length
constexpr int kRows = 16;    // batch rows per wave (WMMA M)

constexpr float kL2E = 1.4426950408889634f;  // log2(e)

typedef __attribute__((ext_vector_type(16))) _Float16 v16h;
typedef __attribute__((ext_vector_type(8)))  _Float16 v8h;
typedef __attribute__((ext_vector_type(2)))  _Float16 v2h;
typedef __attribute__((ext_vector_type(8)))  float    v8f;
typedef __attribute__((ext_vector_type(4)))  float    v4f;

__device__ __forceinline__ float fexp2(float x) { return __builtin_amdgcn_exp2f(x); }
__device__ __forceinline__ float frcp(float x)  { return __builtin_amdgcn_rcpf(x); }

// sigmoid(x + b) with c = -log2e*b precomputed: fma + exp2 + add + rcp
__device__ __forceinline__ float sigm_b(float x, float c) {
  return frcp(1.0f + fexp2(__builtin_fmaf(-kL2E, x, c)));
}
// tanh(x + b) with c = 2*log2e*b precomputed
__device__ __forceinline__ float tanh_b(float x, float c) {
  return 1.0f - 2.0f * frcp(1.0f + fexp2(__builtin_fmaf(2.0f * kL2E, x, c)));
}
// tanh(x) (no bias), for the cell-state nonlinearity
__device__ __forceinline__ float tanh_fast(float x) {
  return 1.0f - 2.0f * frcp(1.0f + fexp2(2.8853900817779268f * x));
}

__device__ __forceinline__ v8f zero8() {
  v8f r;
  #pragma unroll
  for (int e = 0; e < 8; ++e) r[e] = 0.0f;
  return r;
}

// Load x tile for one step and convert straight to the f16 A-operand layout
// (natural K order): halves 0-7 = cols kb..kb+7, halves 8-15 = cols kb+16..kb+23.
__device__ __forceinline__ v16h load_x_a(const float* __restrict__ p, int kb) {
  const v4f a0 = *(const v4f*)(p + kb);
  const v4f a1 = *(const v4f*)(p + kb + 4);
  const v4f b0 = *(const v4f*)(p + kb + 16);
  const v4f b1 = *(const v4f*)(p + kb + 20);
  v16h r;
  #pragma unroll
  for (int e = 0; e < 4; ++e) {
    r[e]      = (_Float16)a0[e];
    r[e + 4]  = (_Float16)a1[e];
    r[e + 8]  = (_Float16)b0[e];
    r[e + 12] = (_Float16)b1[e];
  }
  return r;
}

// Fused masked-LSTM + mean-pool + classifier. One wave (32 lanes) owns 16
// batch rows for the whole sequence; all matmul work is v_wmma_f32_16x16x32_f16
// with both weight matrices resident in VGPRs (128 regs) for the whole scan.
//
// h recurrent operand uses a permuted K order, col(j) = kb + (j>>1) + 16*(j&1),
// so the C-layout -> A-layout transpose is 8 packed ds_store_b32 plus one
// contiguous 32B reload per lane; the W_hh B-tiles are packed with the same
// K permutation so the WMMA contraction is unchanged.
__global__ __launch_bounds__(32) void lstm_fused_kernel(
    const float* __restrict__ x,      // [B, T, H]
    const float* __restrict__ mask,   // [B, T]
    const float* __restrict__ w_ih,   // [4H, H]
    const float* __restrict__ w_hh,   // [4H, H]
    const float* __restrict__ b_ih,   // [4H]
    const float* __restrict__ b_hh,   // [4H]
    const float* __restrict__ w_cls,  // [1, H]
    const float* __restrict__ b_cls,  // [1]
    float* __restrict__ out_pooled,   // [B, H]
    float* __restrict__ out_probs,    // [B]
    int batch)
{
  // h transpose staging (single buffer: one wave + in-order LDS + explicit
  // s_wait_dscnt make double buffering unnecessary). Row stride 40 halves
  // (80B) keeps stores and the contiguous reload bank-conflict-free.
  __shared__ __align__(16) _Float16 sH[kRows * 40];

  const int lane = threadIdx.x;        // 0..31
  const int rsub = lane & 15;          // A tile row / C tile column N
  const int kb   = (lane >> 4) * 8;    // A/B operand K-chunk base for this lane
  const int r0   = blockIdx.x * kRows;
  if (r0 >= batch) return;

  // ---- one-time: pack f16 B-operand weight tiles straight into registers.
  // B element (k, n) of tile nt = w[nt*16 + n, k]  (g = v @ W^T).
  // W_ih uses the natural hardware K order; W_hh uses the permuted order
  // matching the interleaved h reload.
  v16h Wih[8], Whh[8];
  #pragma unroll
  for (int nt = 0; nt < 8; ++nt) {
    const int j = nt * 16 + rsub;
    #pragma unroll
    for (int e = 0; e < 16; ++e) {
      const int kih = kb + e + ((e >= 8) ? 8 : 0);
      const int khh = kb + (e >> 1) + 16 * (e & 1);
      Wih[nt][e] = (_Float16)w_ih[j * kH + kih];
      Whh[nt][e] = (_Float16)w_hh[j * kH + khh];
    }
  }
  // Pre-scaled bias constants, folded into the gate nonlinearity fma:
  //   sigmoid tiles (i,f,o):  c = -log2e * b
  //   tanh tiles (g: 4,5):    c = 2*log2e * b
  float bc[8];
  #pragma unroll
  for (int nt = 0; nt < 8; ++nt) {
    const float b = b_ih[nt * 16 + rsub] + b_hh[nt * 16 + rsub];
    bc[nt] = (nt == 4 || nt == 5) ? (2.0f * kL2E * b) : (-kL2E * b);
  }

  // ---- state ---------------------------------------------------------------
  v8f  cs0, cs1;            // cell state, C-layout tiles (cols 0-15 / 16-31)
  v16h ah;                  // h_{t-1} as f16 A-operand (permuted K order)
  float pooled[16];         // masked sum of h (permuted col order per lane)
  #pragma unroll
  for (int e = 0; e < 8; ++e) { cs0[e] = 0.0f; cs1[e] = 0.0f; }
  #pragma unroll
  for (int e = 0; e < 16; ++e) { ah[e] = (_Float16)0.0f; pooled[e] = 0.0f; }
  float den = 0.0f;

  const float* xrow = x    + (size_t)(r0 + rsub) * kT * kH;
  const float* mrow = mask + (size_t)(r0 + rsub) * kT;

  // software pipeline: x tile for t=0, already converted to the f16 A operand
  v16h axn    = load_x_a(xrow, kb);
  float mnext = mrow[0];

  #pragma unroll 1
  for (int t = 0; t < kT; ++t) {
    const v16h axv = axn;
    const float mt = mnext;

    // deep prefetch of the streaming x read (~8 steps ahead)
    {
      const int tp = (t + 8 < kT) ? t + 8 : kT - 1;
      __builtin_prefetch(xrow + (size_t)tp * kH + kb, 0, 1);
    }
    // pipeline next step's x tile + mask
    {
      const int tn = (t + 1 < kT) ? t + 1 : kT - 1;
      axn   = load_x_a(xrow + (size_t)tn * kH, kb);
      mnext = mrow[tn];
    }

    v8f h0, h1;
    // ---- half 0: hidden cols 0-15 (gate tiles 0/2/4/6) --------------------
    {
      v8f ai = __builtin_amdgcn_wmma_f32_16x16x32_f16(false, axv, false, Wih[0], (short)0, zero8(), false, false);
      v8f af = __builtin_amdgcn_wmma_f32_16x16x32_f16(false, axv, false, Wih[2], (short)0, zero8(), false, false);
      v8f ag = __builtin_amdgcn_wmma_f32_16x16x32_f16(false, axv, false, Wih[4], (short)0, zero8(), false, false);
      v8f ao = __builtin_amdgcn_wmma_f32_16x16x32_f16(false, axv, false, Wih[6], (short)0, zero8(), false, false);
      ai = __builtin_amdgcn_wmma_f32_16x16x32_f16(false, ah, false, Whh[0], (short)0, ai, false, false);
      af = __builtin_amdgcn_wmma_f32_16x16x32_f16(false, ah, false, Whh[2], (short)0, af, false, false);
      ag = __builtin_amdgcn_wmma_f32_16x16x32_f16(false, ah, false, Whh[4], (short)0, ag, false, false);
      ao = __builtin_amdgcn_wmma_f32_16x16x32_f16(false, ah, false, Whh[6], (short)0, ao, false, false);
      #pragma unroll
      for (int e = 0; e < 8; ++e) {
        const float iv = sigm_b(ai[e], bc[0]), fv = sigm_b(af[e], bc[2]);
        const float gv = tanh_b(ag[e], bc[4]), ov = sigm_b(ao[e], bc[6]);
        cs0[e] = fv * cs0[e] + iv * gv;
        h0[e]  = ov * tanh_fast(cs0[e]);
      }
    }
    // ---- half 1: hidden cols 16-31 (gate tiles 1/3/5/7) -------------------
    {
      v8f ai = __builtin_amdgcn_wmma_f32_16x16x32_f16(false, axv, false, Wih[1], (short)0, zero8(), false, false);
      v8f af = __builtin_amdgcn_wmma_f32_16x16x32_f16(false, axv, false, Wih[3], (short)0, zero8(), false, false);
      v8f ag = __builtin_amdgcn_wmma_f32_16x16x32_f16(false, axv, false, Wih[5], (short)0, zero8(), false, false);
      v8f ao = __builtin_amdgcn_wmma_f32_16x16x32_f16(false, axv, false, Wih[7], (short)0, zero8(), false, false);
      ai = __builtin_amdgcn_wmma_f32_16x16x32_f16(false, ah, false, Whh[1], (short)0, ai, false, false);
      af = __builtin_amdgcn_wmma_f32_16x16x32_f16(false, ah, false, Whh[3], (short)0, af, false, false);
      ag = __builtin_amdgcn_wmma_f32_16x16x32_f16(false, ah, false, Whh[5], (short)0, ag, false, false);
      ao = __builtin_amdgcn_wmma_f32_16x16x32_f16(false, ah, false, Whh[7], (short)0, ao, false, false);
      #pragma unroll
      for (int e = 0; e < 8; ++e) {
        const float iv = sigm_b(ai[e], bc[1]), fv = sigm_b(af[e], bc[3]);
        const float gv = tanh_b(ag[e], bc[5]), ov = sigm_b(ao[e], bc[7]);
        cs1[e] = fv * cs1[e] + iv * gv;
        h1[e]  = ov * tanh_fast(cs1[e]);
      }
    }

    // transpose h via LDS, packed-pair layout: dword d of row r holds
    // (col d, col d+16) -> 8 x v_cvt_pk_f16_f32 + 8 x ds_store_b32,
    // consecutive lanes hit consecutive banks.
    {
      const int crow0 = (lane >> 4) * 8;  // C-layout rows held by this lane
      #pragma unroll
      for (int e = 0; e < 8; ++e) {
        v2h p;
        p[0] = (_Float16)h0[e];
        p[1] = (_Float16)h1[e];
        *(v2h*)(sH + (crow0 + e) * 40 + rsub * 2) = p;
      }
    }
    // single-wave handoff: LDS pipe is in-order; drain stores + pin ordering
    asm volatile("s_wait_dscnt 0x0" ::: "memory");
    {
      // contiguous 32B per lane: dwords kb..kb+7 of row rsub
      const v8h a = *(const v8h*)(sH + rsub * 40 + kb * 2);
      const v8h b = *(const v8h*)(sH + rsub * 40 + kb * 2 + 8);
      #pragma unroll
      for (int e = 0; e < 8; ++e) { ah[e] = a[e]; ah[e + 8] = b[e]; }
    }

    // masked mean-pool accumulation (one mask value per lane;
    // float(f16)*f32+f32 pattern-matches to v_fma_mix_f32)
    den += mt;
    #pragma unroll
    for (int e = 0; e < 16; ++e) pooled[e] += mt * (float)ah[e];
  }

  // ---- finalize: pooled mean + classifier ---------------------------------
  // pooled[j] holds column kb + (j>>1) + 16*(j&1)
  const float rden = frcp(fmaxf(den, 1e-8f));
  const int r = r0 + rsub;
  float* pr = out_pooled + (size_t)r * kH;
  v4f lo0, lo1, hi0, hi1;
  #pragma unroll
  for (int d = 0; d < 4; ++d) {
    lo0[d] = pooled[2 * d]           * rden;  // cols kb+0..kb+3
    lo1[d] = pooled[2 * (d + 4)]     * rden;  // cols kb+4..kb+7
    hi0[d] = pooled[2 * d + 1]       * rden;  // cols kb+16..kb+19
    hi1[d] = pooled[2 * (d + 4) + 1] * rden;  // cols kb+20..kb+23
  }
  *(v4f*)(pr + kb)      = lo0;
  *(v4f*)(pr + kb + 4)  = lo1;
  *(v4f*)(pr + kb + 16) = hi0;
  *(v4f*)(pr + kb + 20) = hi1;

  float s = 0.0f;
  #pragma unroll
  for (int e = 0; e < 16; ++e) {
    const int col = kb + (e >> 1) + 16 * (e & 1);
    s += (pooled[e] * rden) * w_cls[col];
  }
  s += __shfl_xor(s, 16, 32);  // combine the two half-row lanes (wave32)
  if (lane < 16) out_probs[r] = frcp(1.0f + fexp2(-kL2E * (s + b_cls[0])));
}

}  // namespace

extern "C" void kernel_launch(void* const* d_in, const int* in_sizes, int n_in,
                              void* d_out, int out_size, void* d_ws, size_t ws_size,
                              hipStream_t stream) {
  (void)n_in; (void)out_size; (void)d_ws; (void)ws_size;
  const float* x     = (const float*)d_in[0];
  const float* mask  = (const float*)d_in[1];
  const float* w_ih  = (const float*)d_in[2];
  const float* w_hh  = (const float*)d_in[3];
  const float* b_ih  = (const float*)d_in[4];
  const float* b_hh  = (const float*)d_in[5];
  const float* w_cls = (const float*)d_in[6];
  const float* b_cls = (const float*)d_in[7];

  const int batch = in_sizes[1] / kT;  // mask is [B, T]
  float* out_pooled = (float*)d_out;                          // [B, H]
  float* out_probs  = (float*)d_out + (size_t)batch * kH;     // [B]

  dim3 grid((batch + kRows - 1) / kRows), block(32);
  lstm_fused_kernel<<<grid, block, 0, stream>>>(
      x, mask, w_ih, w_hh, b_ih, b_hh, w_cls, b_cls,
      out_pooled, out_probs, batch);
}
